// MambaAdapterStack_5755256176891
// MI455X (gfx1250) — matrix operand order repaired and checked by reading, not compiled, not run on hardware
//
#include <hip/hip_runtime.h>

typedef __attribute__((ext_vector_type(16))) __bf16 v16bf;
typedef __attribute__((ext_vector_type(8)))  float  v8f;
typedef int v4i __attribute__((vector_size(16)));

union FragU { v16bf v; uint4 u[2]; };

#if defined(__has_builtin)
#if __has_builtin(__builtin_amdgcn_global_load_async_to_lds_b128) && \
    __has_builtin(__builtin_amdgcn_s_wait_asynccnt)
#define USE_ASYNC_LDS 1
#endif
#endif
#ifndef USE_ASYNC_LDS
#define USE_ASYNC_LDS 0
#endif

#define AS_GLOBAL __attribute__((address_space(1)))
#define AS_LDS    __attribute__((address_space(3)))

// ---------------------------------------------------------------------------
// Generic NT GEMM: out[m,n] = sum_k A[m,k] * W[n,k]   (A: MxK bf16, W: NxK bf16)
// Block tile 128x64, BK=32. 8 waves, each wave -> 32x32 (2x2 wmma tiles).
// Double-buffered LDS; tiles staged with GLOBAL_LOAD_ASYNC_TO_LDS_B128 and
// pipelined against the WMMAs via ASYNCcnt (sync fallback if builtin absent).
// MODE 0: f32 out            MODE 1: softplus(acc + bias[n]) f32 out
// MODE 2: bf16 out (optional row reversal within length Lrev, col offset)
// MODE 3: f32 out = resid + acc + bias[n]
// ---------------------------------------------------------------------------
template<int MODE>
__global__ __launch_bounds__(256)
void gemm_bf16(const __bf16* __restrict__ A, const __bf16* __restrict__ W,
               int M, int N, int Kd,
               const float* __restrict__ bias, const float* __restrict__ resid,
               float* __restrict__ outF, __bf16* __restrict__ outB,
               int ldo, int colOff, int Lrev)
{
  const int tid  = threadIdx.x;
  const int lane = tid & 31;
  const int w    = tid >> 5;      // 0..7
  const int wm   = w & 3;         // 4 row groups of 32
  const int wn   = w >> 2;        // 2 col groups of 32
  const int m0   = blockIdx.x * 128;
  const int n0   = blockIdx.y * 64;

  __shared__ __bf16 sA[2][128 * 40]; // pad row stride to 40 elems (80B, 16B aligned)
  __shared__ __bf16 sB[2][64 * 40];

  v8f acc[2][2] = {};

  const int  ldrow = tid >> 2;          // 0..63
  const int  seg   = (tid & 3) * 8;     // 0,8,16,24 elems
  const int  nB    = n0 + ldrow;
  const bool fullB = (n0 + 64 <= N);

  auto issue = [&](int k0, int buf) {
#if USE_ASYNC_LDS
    #pragma unroll
    for (int p = 0; p < 2; ++p) {
      const int row = ldrow + p * 64;
      __builtin_amdgcn_global_load_async_to_lds_b128(
          (AS_GLOBAL v4i*)(A + (size_t)(m0 + row) * Kd + k0 + seg),
          (AS_LDS v4i*)&sA[buf][row * 40 + seg], 0, 0);
    }
    if (fullB) {
      __builtin_amdgcn_global_load_async_to_lds_b128(
          (AS_GLOBAL v4i*)(W + (size_t)nB * Kd + k0 + seg),
          (AS_LDS v4i*)&sB[buf][ldrow * 40 + seg], 0, 0);
    } else {
      uint4 vb = make_uint4(0u, 0u, 0u, 0u);
      if (nB < N) vb = *(const uint4*)(W + (size_t)nB * Kd + k0 + seg);
      *(uint4*)(&sB[buf][ldrow * 40 + seg]) = vb;
    }
#else
    if (k0 + 32 < Kd) __builtin_prefetch(A + (size_t)(m0 + ldrow) * Kd + k0 + 32, 0, 3);
    #pragma unroll
    for (int p = 0; p < 2; ++p) {
      const int row = ldrow + p * 64;
      uint4 v = *(const uint4*)(A + (size_t)(m0 + row) * Kd + k0 + seg);
      *(uint4*)(&sA[buf][row * 40 + seg]) = v;
    }
    uint4 vb = make_uint4(0u, 0u, 0u, 0u);
    if (nB < N) vb = *(const uint4*)(W + (size_t)nB * Kd + k0 + seg);
    *(uint4*)(&sB[buf][ldrow * 40 + seg]) = vb;
#endif
  };

  const int nk = Kd >> 5;
  issue(0, 0);
  for (int it = 0; it < nk; ++it) {
    const int  buf     = it & 1;
    const bool hasNext = (it + 1 < nk);
    if (hasNext) issue((it + 1) << 5, buf ^ 1); // buf^1 free since tail barrier of it-1
#if USE_ASYNC_LDS
    // async loads complete in order: <=3 (or <=2) outstanding => group `it` landed
    if (hasNext) {
      if (fullB) __builtin_amdgcn_s_wait_asynccnt(3);
      else       __builtin_amdgcn_s_wait_asynccnt(2);
    } else {
      __builtin_amdgcn_s_wait_asynccnt(0);
    }
#endif
    __syncthreads();

    // A fragment (16-bit 16x32): lane<16 holds K {0..7,16..23}, lane>=16 {8..15,24..31}
    const int fr   = lane & 15;
    const int koff = (lane >> 4) << 3;       // 0 or 8
    const int kbb  = (lane >> 4) << 4;       // 0 or 16
    FragU a[2], b[2];
    #pragma unroll
    for (int mi = 0; mi < 2; ++mi) {
      const __bf16* p = &sA[buf][(wm * 32 + mi * 16 + fr) * 40 + koff];
      a[mi].u[0] = *(const uint4*)(p);
      a[mi].u[1] = *(const uint4*)(p + 16);
    }
    // B fragment (32x16): lane<16 holds K {0..7,16..23} of col (lane&15), lane>=16 {8..15,24..31}
    #pragma unroll
    for (int ni = 0; ni < 2; ++ni) {
      const __bf16* p = &sB[buf][(wn * 32 + ni * 16 + fr) * 40 + koff];
      b[ni].u[0] = *(const uint4*)(p);
      b[ni].u[1] = *(const uint4*)(p + 16);
    }
    #pragma unroll
    for (int mi = 0; mi < 2; ++mi)
      #pragma unroll
      for (int ni = 0; ni < 2; ++ni)
        acc[mi][ni] = __builtin_amdgcn_wmma_f32_16x16x32_bf16(
            false, a[mi].v, false, b[ni].v, (short)0, acc[mi][ni], false, false);
    __syncthreads();
  }

  // epilogue: C/D layout -> VGPR r: lanes 0-15 M=r, lanes 16-31 M=r+8; N = lane&15
  #pragma unroll
  for (int mi = 0; mi < 2; ++mi) {
    #pragma unroll
    for (int ni = 0; ni < 2; ++ni) {
      const int nb = n0 + wn * 32 + ni * 16 + (lane & 15);
      if (nb >= N) continue;
      const int mb = m0 + wm * 32 + mi * 16 + ((lane >> 4) << 3);
      #pragma unroll
      for (int r = 0; r < 8; ++r) {
        const int m = mb + r;
        const float v = acc[mi][ni][r];
        if (MODE == 0) {
          outF[(size_t)m * ldo + colOff + nb] = v;
        } else if (MODE == 1) {
          float xv = v + bias[nb];
          outF[(size_t)m * ldo + colOff + nb] = (xv > 20.f) ? xv : __logf(1.f + __expf(xv));
        } else if (MODE == 2) {
          int dm = m;
          if (Lrev) { int ll = m & (Lrev - 1); dm = m - ll + (Lrev - 1 - ll); }
          outB[(size_t)dm * ldo + colOff + nb] = (__bf16)v;
        } else { // MODE 3
          size_t idx = (size_t)m * ldo + colOff + nb;
          outF[idx] = resid[idx] + v + bias[nb];
        }
      }
    }
  }
}

// ---------------------------------------------------------------------------
// LayerNorm over rows of D=1024. mode 0: bf16 out + bf16 reversed-out.
// mode 1: f32 out.
// ---------------------------------------------------------------------------
__global__ __launch_bounds__(256)
void layernorm_k(const float* __restrict__ x, const float* __restrict__ w,
                 const float* __restrict__ b, __bf16* __restrict__ obf,
                 __bf16* __restrict__ obfrev, float* __restrict__ of,
                 int D, int L, int mode)
{
  const int row = blockIdx.x;
  const int tid = threadIdx.x;
  __shared__ float r1[256], r2[256];
  const float* xr = x + (size_t)row * D;
  float v[4]; float s = 0.f, s2 = 0.f;
  #pragma unroll
  for (int i = 0; i < 4; ++i) { float t = xr[tid + i * 256]; v[i] = t; s += t; s2 += t * t; }
  r1[tid] = s; r2[tid] = s2; __syncthreads();
  for (int off = 128; off > 0; off >>= 1) {
    if (tid < off) { r1[tid] += r1[tid + off]; r2[tid] += r2[tid + off]; }
    __syncthreads();
  }
  const float mu = r1[0] / D;
  const float var = r2[0] / D - mu * mu;
  const float rs = rsqrtf(var + 1e-5f);
  const int l = row & (L - 1);
  const int rrow = row - l + (L - 1 - l);
  #pragma unroll
  for (int i = 0; i < 4; ++i) {
    int c = tid + i * 256;
    float y = (v[i] - mu) * rs * w[c] + b[c];
    if (mode == 0) {
      obf[(size_t)row * D + c]     = (__bf16)y;
      obfrev[(size_t)rrow * D + c] = (__bf16)y;
    } else {
      of[(size_t)row * D + c] = y;
    }
  }
}

// depthwise causal conv (K=4) + SiLU; u lives in xz[:, 0:DI]
__global__ __launch_bounds__(256)
void conv_silu_k(const __bf16* __restrict__ xz, const float* __restrict__ cw,
                 const float* __restrict__ cb, float* __restrict__ ucf,
                 __bf16* __restrict__ ucb, int L, int DI)
{
  const size_t i = (size_t)blockIdx.x * 256 + threadIdx.x;
  const int d = (int)(i % DI);
  const size_t m = i / DI;
  const int l = (int)(m & (size_t)(L - 1));
  float acc = cb[d];
  #pragma unroll
  for (int k = 0; k < 4; ++k) {
    int ls = l + k - 3;
    if (ls >= 0)
      acc += cw[d * 4 + k] * (float)xz[(m + (size_t)(ls - l)) * (size_t)(2 * DI) + d];
  }
  float sv = acc / (1.f + __expf(-acc)); // silu
  ucf[i] = sv; ucb[i] = (__bf16)sv;
}

__global__ void cvt_k(const float* __restrict__ s, __bf16* __restrict__ d, size_t n)
{
  for (size_t i = (size_t)blockIdx.x * blockDim.x + threadIdx.x; i < n;
       i += (size_t)gridDim.x * blockDim.x)
    d[i] = (__bf16)s[i];
}

__global__ void delta_cvt_k(const float* __restrict__ xdbl, __bf16* __restrict__ dbf, size_t n)
{
  size_t i = (size_t)blockIdx.x * blockDim.x + threadIdx.x;
  if (i < n) { size_t m = i >> 6; int j = (int)(i & 63); dbf[i] = (__bf16)xdbl[m * 96 + j]; }
}

// ---------------------------------------------------------------------------
// Selective scan: one thread per (b,d); 16-state recurrence in registers.
// B/C (xdbl cols 64..95) staged in LDS per 128-step chunk. Fuses Dsk*uc and
// silu(z) gate; writes bf16 y for the out GEMM.
// ---------------------------------------------------------------------------
__global__ __launch_bounds__(256)
void scan_k(const float* __restrict__ dt, const float* __restrict__ ucf,
            const float* __restrict__ xdbl, const __bf16* __restrict__ xz,
            const float* __restrict__ A_log, const float* __restrict__ Dsk,
            __bf16* __restrict__ ybf, int L, int DI)
{
  const int tid = threadIdx.x;
  const int chunks = DI / 256;
  const int bb = blockIdx.x / chunks;
  const int d  = (blockIdx.x % chunks) * 256 + tid;
  float Aa[16], h[16];
  #pragma unroll
  for (int n = 0; n < 16; ++n) { Aa[n] = -__expf(A_log[(size_t)d * 16 + n]); h[n] = 0.f; }
  const float dsk = Dsk[d];
  __shared__ float sBC[128][32];
  for (int l0 = 0; l0 < L; l0 += 128) {
    __syncthreads();
    for (int i = tid; i < 128 * 32; i += 256) {
      int st = i >> 5, c = i & 31;
      sBC[st][c] = xdbl[((size_t)(bb * L + l0 + st)) * 96 + 64 + c];
    }
    __syncthreads();
    for (int s = 0; s < 128; ++s) {
      const size_t row = (size_t)(bb * L + l0 + s);
      const size_t idx = row * (size_t)DI + d;
      const float dtv = dt[idx];
      const float uv  = ucf[idx];
      const float du  = dtv * uv;
      float y = 0.f;
      #pragma unroll
      for (int n = 0; n < 16; ++n) {
        const float dA = __expf(dtv * Aa[n]);
        h[n] = dA * h[n] + du * sBC[s][n];
        y += h[n] * sBC[s][16 + n];
      }
      const float zv = (float)xz[row * (size_t)(2 * DI) + DI + d];
      const float yo = (y + dsk * uv) * (zv / (1.f + __expf(-zv)));
      ybf[idx] = (__bf16)yo;
    }
  }
}

// ---------------------------------------------------------------------------
extern "C" void kernel_launch(void* const* d_in, const int* in_sizes, int n_in,
                              void* d_out, int out_size, void* d_ws, size_t ws_size,
                              hipStream_t stream)
{
  (void)in_sizes; (void)n_in; (void)out_size; (void)ws_size;
  constexpr int Bk = 4, Lk = 2048, DM = 1024, DI = 2048;
  constexpr size_t MT = (size_t)Bk * Lk; // 8192 rows

  const float* x      = (const float*)d_in[0];
  const float* norm_w = (const float*)d_in[1];
  const float* norm_b = (const float*)d_in[2];
  const float* in_w   = (const float*)d_in[3];
  const float* conv_w = (const float*)d_in[4];
  const float* conv_b = (const float*)d_in[5];
  const float* xp_w   = (const float*)d_in[6];
  const float* dt_w   = (const float*)d_in[7];
  const float* dt_b   = (const float*)d_in[8];
  const float* A_log  = (const float*)d_in[9];
  const float* Dsk    = (const float*)d_in[10];
  const float* out_w  = (const float*)d_in[11];
  const float* proj_w = (const float*)d_in[12];
  const float* proj_b = (const float*)d_in[13];
  const float* fnorm_w = (const float*)d_in[14];
  const float* fnorm_b = (const float*)d_in[15];

  char* ws = (char*)d_ws;
  size_t off = 0;
  auto alloc = [&](size_t bytes) -> char* {
    char* p = ws + off; off += (bytes + 255) & ~(size_t)255; return p;
  };

  __bf16* hbf     = (__bf16*)alloc(MT * DM * 2);
  __bf16* hrevbf  = (__bf16*)alloc(MT * DM * 2);
  __bf16* xzb     = (__bf16*)alloc(MT * (size_t)(2 * DI) * 2);
  float*  ucf     = (float*) alloc(MT * DI * 4);
  __bf16* ucb     = (__bf16*)alloc(MT * DI * 2);
  float*  xdbl    = (float*) alloc(MT * 96 * 4);
  __bf16* dbf     = (__bf16*)alloc(MT * 64 * 2);
  float*  dtbuf   = (float*) alloc(MT * DI * 4);
  __bf16* ybf     = (__bf16*)alloc(MT * DI * 2);
  __bf16* combbf  = (__bf16*)alloc(MT * (size_t)(2 * DM) * 2);
  float*  hs      = (float*) alloc(MT * DM * 4);
  __bf16* w_in    = (__bf16*)alloc((size_t)4096 * 1024 * 2);
  __bf16* w_xp    = (__bf16*)alloc((size_t)96 * 2048 * 2);
  __bf16* w_dt    = (__bf16*)alloc((size_t)2048 * 64 * 2);
  __bf16* w_out   = (__bf16*)alloc((size_t)1024 * 2048 * 2);
  __bf16* w_proj  = (__bf16*)alloc((size_t)1024 * 2048 * 2);

  (void)hipMemcpyAsync(hs, x, MT * DM * sizeof(float), hipMemcpyDeviceToDevice, stream);

  for (int l = 0; l < 2; ++l) {
    layernorm_k<<<(int)MT, 256, 0, stream>>>(hs, norm_w + l * DM, norm_b + l * DM,
                                             hbf, hrevbf, nullptr, DM, Lk, 0);
    for (int dir = 0; dir < 2; ++dir) {
      const int slc = l * 2 + dir;
      // in_proj: (8192x1024)x(4096x1024)^T -> xz bf16
      cvt_k<<<2048, 256, 0, stream>>>(in_w + (size_t)slc * 4096 * 1024, w_in,
                                      (size_t)4096 * 1024);
      gemm_bf16<2><<<dim3(64, 64), 256, 0, stream>>>(
          dir ? hrevbf : hbf, w_in, 8192, 4096, 1024,
          nullptr, nullptr, nullptr, xzb, 2 * DI, 0, 0);
      // depthwise conv + silu
      conv_silu_k<<<(int)(MT * DI / 256), 256, 0, stream>>>(
          xzb, conv_w + (size_t)slc * DI * 4, conv_b + (size_t)slc * DI, ucf, ucb, Lk, DI);
      // x_proj: (8192x2048)x(96x2048)^T -> xdbl f32
      cvt_k<<<256, 256, 0, stream>>>(xp_w + (size_t)slc * 96 * 2048, w_xp, (size_t)96 * 2048);
      gemm_bf16<0><<<dim3(64, 2), 256, 0, stream>>>(
          ucb, w_xp, 8192, 96, 2048, nullptr, nullptr, xdbl, nullptr, 96, 0, 0);
      // dt: softplus((8192x64)x(2048x64)^T + dt_b)
      delta_cvt_k<<<(int)(MT * 64 / 256), 256, 0, stream>>>(xdbl, dbf, MT * 64);
      cvt_k<<<128, 256, 0, stream>>>(dt_w + (size_t)slc * 2048 * 64, w_dt, (size_t)2048 * 64);
      gemm_bf16<1><<<dim3(64, 32), 256, 0, stream>>>(
          dbf, w_dt, 8192, 2048, 64, dt_b + (size_t)slc * DI, nullptr, dtbuf, nullptr,
          DI, 0, 0);
      // selective scan (+Dsk*uc, *silu(z)) -> y bf16
      scan_k<<<dim3(Bk * (DI / 256)), 256, 0, stream>>>(
          dtbuf, ucf, xdbl, xzb, A_log + (size_t)slc * DI * 16, Dsk + (size_t)slc * DI,
          ybf, Lk, DI);
      // out_proj: (8192x2048)x(1024x2048)^T -> combined (bwd rows un-reversed)
      cvt_k<<<2048, 256, 0, stream>>>(out_w + (size_t)slc * 1024 * 2048, w_out,
                                      (size_t)1024 * 2048);
      gemm_bf16<2><<<dim3(64, 16), 256, 0, stream>>>(
          ybf, w_out, 8192, 1024, 2048, nullptr, nullptr, nullptr, combbf,
          2 * DM, dir * DM, dir ? Lk : 0);
    }
    // residual proj: hs = hs + combined x proj_w^T + proj_b
    cvt_k<<<2048, 256, 0, stream>>>(proj_w + (size_t)l * 1024 * 2048, w_proj,
                                    (size_t)1024 * 2048);
    gemm_bf16<3><<<dim3(64, 16), 256, 0, stream>>>(
        combbf, w_proj, 8192, 1024, 2048, proj_b + (size_t)l * DM, hs, hs, nullptr,
        DM, 0, 0);
  }
  layernorm_k<<<(int)MT, 256, 0, stream>>>(hs, fnorm_w, fnorm_b, nullptr, nullptr,
                                           (float*)d_out, DM, Lk, 1);
}
